// NeuralMapModel_83451214561582
// MI455X (gfx1250) — matrix operand order repaired and checked by reading, not compiled, not run on hardware
//
#include <hip/hip_runtime.h>
#include <hip/hip_bf16.h>

typedef __attribute__((ext_vector_type(16))) _Float16 v16h;
typedef __attribute__((ext_vector_type(8)))  _Float16 v8h;
typedef __attribute__((ext_vector_type(8)))  float    v8f;

#define CDIM 200
#define FDIM 2000
#define HIDN 64
#define ODIM 25
#define LROW 72   // LDS row stride in halfs: 144B, multiple of 16B

static __device__ __forceinline__ v8f wmma16(v16h a, v16h b, v8f c) {
  return __builtin_amdgcn_wmma_f32_16x16x32_f16(false, a, false, b, (short)0, c,
                                                false, false);
}

// Bilinear sample of 8 consecutive feature channels [fbase, fbase+8) at (xi, yi).
// x0 is clamped to dim-2 (with wx recomputed) so x1 == x0+1 always; this is
// exactly equivalent to the reference clamp semantics and lets each corner
// pair be one 8-byte load.
__device__ __forceinline__ void bilerp8(const float* __restrict__ plane,
                                        float xi, float yi, int dim, int fbase,
                                        float* r) {
  float fd = (float)(dim - 1);
  float x = fminf(fmaxf(xi, 0.f), fd);
  float y = fminf(fmaxf(yi, 0.f), fd);
  int x0 = min((int)floorf(x), dim - 2);
  int y0 = min((int)floorf(y), dim - 2);
  float wx = x - (float)x0;
  float wy = y - (float)y0;
  float wx1 = 1.f - wx, wy1 = 1.f - wy;
  int i0 = y0 * dim + x0;      // row y0, columns x0 and x0+1
  int i1 = i0 + dim;           // row y0+1
  const float* pb = plane + (size_t)fbase * dim * dim;
#pragma unroll
  for (int f = 0; f < 8; ++f) {
    const float* q = pb + (size_t)f * dim * dim;
    float2 r0 = *(const float2*)(q + i0);
    float2 r1 = *(const float2*)(q + i1);
    r[f] = (r0.x * wx1 + r0.y * wx) * wy1 + (r1.x * wx1 + r1.y * wx) * wy;
  }
}

__global__ __launch_bounds__(256) void neural_map_wmma(
    const float* __restrict__ coords,
    const float* __restrict__ cplane,
    const float* __restrict__ fplane,
    const float* __restrict__ W1, const float* __restrict__ b1,
    const float* __restrict__ W2, const float* __restrict__ b2,
    const float* __restrict__ Wo, const float* __restrict__ bo,
    float* __restrict__ out, int npts)
{
  __shared__ __align__(16) _Float16 lds[8][16 * LROW];
  const int lane = threadIdx.x & 31;
  const int hi   = lane >> 4;   // half-wave id (A/B layout K-offset selector)
  const int ln   = lane & 15;   // row (point) / column index within tile
  _Float16* hbuf = lds[threadIdx.x >> 5];

  // ---- Preload weights into WMMA B layout (f16), once per wave ----
  // B layout: lane holds column N = t*16+ln, element e holds K = e + 8*hi + (e & 8).
  v16h w1b[4];
#pragma unroll
  for (int t = 0; t < 4; ++t)
#pragma unroll
    for (int e = 0; e < 16; ++e)
      w1b[t][e] = (_Float16)W1[(e + 8 * hi + (e & 8)) * HIDN + t * 16 + ln];

  v16h w2b[2][4];
#pragma unroll
  for (int kc = 0; kc < 2; ++kc)
#pragma unroll
    for (int t = 0; t < 4; ++t)
#pragma unroll
      for (int e = 0; e < 16; ++e)
        w2b[kc][t][e] =
            (_Float16)W2[(32 * kc + e + 8 * hi + (e & 8)) * HIDN + t * 16 + ln];

  v16h wob[2][2];
#pragma unroll
  for (int kc = 0; kc < 2; ++kc)
#pragma unroll
    for (int t = 0; t < 2; ++t) {
      int Nn = t * 16 + ln;
      float msk = (Nn < ODIM) ? 1.f : 0.f;
      int Nc = (Nn < ODIM) ? Nn : (ODIM - 1);
#pragma unroll
      for (int e = 0; e < 16; ++e)
        wob[kc][t][e] =
            (_Float16)(Wo[(32 * kc + e + 8 * hi + (e & 8)) * ODIM + Nc] * msk);
    }

  float b1v[4], b2v[4], bov[2];
#pragma unroll
  for (int t = 0; t < 4; ++t) {
    b1v[t] = b1[t * 16 + ln];
    b2v[t] = b2[t * 16 + ln];
  }
#pragma unroll
  for (int t = 0; t < 2; ++t) {
    int Nn = t * 16 + ln;
    bov[t] = (Nn < ODIM) ? bo[Nn] : 0.f;
  }

  float* occ = out;
  float* sem = out + npts;
  float* col = out + (size_t)22 * npts;

  const int nTiles = npts >> 4;
  const int waveId = blockIdx.x * (blockDim.x >> 5) + (threadIdx.x >> 5);
  const int nWaves = gridDim.x * (blockDim.x >> 5);
  const float2* cxy = (const float2*)coords;

  for (int tile = waveId; tile < nTiles; tile += nWaves) {
    const int p = tile * 16 + ln;           // point owned by this lane pair
    float2 xy = cxy[p];

    // Each half-wave lane computes 8 coarse + 8 fine channels of its point,
    // which lands exactly in the WMMA 16-bit A layout (K = feature index).
    float cf[8], ff[8];
    bilerp8(cplane, xy.x * 5.0f  + 0.5f * CDIM, xy.y * 5.0f  + 0.5f * CDIM,
            CDIM, hi * 8, cf);
    bilerp8(fplane, xy.x * 50.0f + 0.5f * FDIM, xy.y * 50.0f + 0.5f * FDIM,
            FDIM, hi * 8, ff);

    v16h a1;
#pragma unroll
    for (int e = 0; e < 8; ++e) {
      a1[e]     = (_Float16)cf[e];   // K = e       (+8 if hi)
      a1[8 + e] = (_Float16)ff[e];   // K = 16 + e  (+8 if hi)
    }

    // ---- Layer 1: (16x32) x (32x64) ----
    v8f c1[4];
#pragma unroll
    for (int t = 0; t < 4; ++t) { v8f z = {}; c1[t] = wmma16(a1, w1b[t], z); }

    // ReLU + bias, stash as h[16][64] f16 in per-wave LDS slab.
    // C layout: element j -> row M = j + 8*hi, col N = t*16 + ln.
#pragma unroll
    for (int t = 0; t < 4; ++t)
#pragma unroll
      for (int j = 0; j < 8; ++j) {
        float v = fmaxf(c1[t][j] + b1v[t], 0.f);
        hbuf[(j + 8 * hi) * LROW + t * 16 + ln] = (_Float16)v;
      }

    // Re-gather in A layout: row M = ln, K-runs at 32*kc + {0,16} + 8*hi.
    v16h a2[2];
#pragma unroll
    for (int kc = 0; kc < 2; ++kc) {
      v8h lo = *(const v8h*)(hbuf + ln * LROW + 32 * kc + 8 * hi);
      v8h hv = *(const v8h*)(hbuf + ln * LROW + 32 * kc + 16 + 8 * hi);
#pragma unroll
      for (int e = 0; e < 8; ++e) { a2[kc][e] = lo[e]; a2[kc][8 + e] = hv[e]; }
    }

    // ---- Layer 2: (16x64) x (64x64), two K-chunks accumulated ----
    v8f c2[4];
#pragma unroll
    for (int t = 0; t < 4; ++t) {
      v8f z = {};
      z = wmma16(a2[0], w2b[0][t], z);
      c2[t] = wmma16(a2[1], w2b[1][t], z);
    }

#pragma unroll
    for (int t = 0; t < 4; ++t)
#pragma unroll
      for (int j = 0; j < 8; ++j) {
        float v = fmaxf(c2[t][j] + b2v[t], 0.f);
        hbuf[(j + 8 * hi) * LROW + t * 16 + ln] = (_Float16)v;
      }

    v16h a3[2];
#pragma unroll
    for (int kc = 0; kc < 2; ++kc) {
      v8h lo = *(const v8h*)(hbuf + ln * LROW + 32 * kc + 8 * hi);
      v8h hv = *(const v8h*)(hbuf + ln * LROW + 32 * kc + 16 + 8 * hi);
#pragma unroll
      for (int e = 0; e < 8; ++e) { a3[kc][e] = lo[e]; a3[kc][8 + e] = hv[e]; }
    }

    // ---- Output layer: (16x64) x (64x25 padded to 32) ----
    v8f c3[2];
#pragma unroll
    for (int t = 0; t < 2; ++t) {
      v8f z = {};
      z = wmma16(a3[0], wob[0][t], z);
      c3[t] = wmma16(a3[1], wob[1][t], z);
    }

    // Scatter to the three output tensors. Consecutive lanes write
    // consecutive sem columns -> coalesced within each half-wave.
#pragma unroll
    for (int t = 0; t < 2; ++t) {
      int c = t * 16 + ln;
#pragma unroll
      for (int j = 0; j < 8; ++j) {
        int pp = tile * 16 + j + 8 * hi;
        float v = c3[t][j] + bov[t];
        if (c == 0) {
          occ[pp] = v;
        } else if (c < 22) {
          sem[(size_t)pp * 21 + (c - 1)] = v;
        } else if (c < 25) {
          col[(size_t)pp * 3 + (c - 22)] = 1.f / (1.f + expf(-v));
        }
      }
    }
  }
}

extern "C" void kernel_launch(void* const* d_in, const int* in_sizes, int n_in,
                              void* d_out, int out_size, void* d_ws, size_t ws_size,
                              hipStream_t stream) {
  const float* coords = (const float*)d_in[0];
  const float* cplane = (const float*)d_in[1];
  const float* fplane = (const float*)d_in[2];
  const float* W1     = (const float*)d_in[3];
  const float* b1     = (const float*)d_in[4];
  const float* W2     = (const float*)d_in[5];
  const float* b2     = (const float*)d_in[6];
  const float* Wo     = (const float*)d_in[7];
  const float* bo     = (const float*)d_in[8];
  float* out = (float*)d_out;

  int npts   = in_sizes[0] / 2;          // coords is (N, 2)
  int nTiles = npts / 16;
  int waves_needed = nTiles;             // 1 tile per wave minimum granularity
  int blocks = (waves_needed + 7) / 8;   // 8 waves per 256-thread block
  if (blocks > 2048) blocks = 2048;      // wave-strided loop amortizes weight load
  if (blocks < 1) blocks = 1;

  neural_map_wmma<<<blocks, 256, 0, stream>>>(coords, cplane, fplane,
                                              W1, b1, W2, b2, Wo, bo,
                                              out, npts);
}
